// Edgeweight_SwarmNet_32753420599779
// MI455X (gfx1250) — compile-verified, matrix operand order checked by reading, not run on hardware
//
#include <hip/hip_runtime.h>

typedef __attribute__((ext_vector_type(16))) _Float16 v16h;
typedef __attribute__((ext_vector_type(8)))  float    v8f;

#define HID 128

__device__ __forceinline__ float lrelu(float v) { return v >= 0.f ? v : 0.1f * v; }

// ---------------------------------------------------------------------------
// 1) View encoder: two strided conv1d + lrelu, concat with pose.
//    One block (128 threads) per node. Writes x padded to stride 32 (f32)
//    and zeroes agg for the later scatter.
// ---------------------------------------------------------------------------
__global__ void encoder_kernel(const float* __restrict__ pose,
                               const float* __restrict__ views,
                               const float* __restrict__ w1, const float* __restrict__ b1,
                               const float* __restrict__ w2, const float* __restrict__ b2,
                               float* __restrict__ x, float* __restrict__ agg)
{
    __shared__ float sv[300];   // views row (3x100)
    __shared__ float s1[441];   // conv1 out (9x49)
    const int n = blockIdx.x, t = threadIdx.x;

    for (int i = t; i < 300; i += 128) sv[i] = views[(size_t)n * 300 + i];
    __syncthreads();

    for (int i = t; i < 441; i += 128) {
        int oc = i / 49, p = i % 49;
        float acc = b1[oc];
#pragma unroll
        for (int ic = 0; ic < 3; ++ic)
#pragma unroll
            for (int k = 0; k < 3; ++k)
                acc += sv[ic * 100 + 2 * p + k] * w1[(oc * 3 + ic) * 3 + k];
        s1[i] = lrelu(acc);
    }
    __syncthreads();

    if (t < 32) {
        float val = 0.f;
        if (t < 3) {
            val = pose[n * 3 + t];
        } else if (t < 27) {
            int p = t - 3;
            float acc = b2[0];
            for (int ic = 0; ic < 9; ++ic)
#pragma unroll
                for (int k = 0; k < 3; ++k)
                    acc += s1[ic * 49 + 2 * p + k] * w2[ic * 3 + k];
            val = lrelu(acc);
        }
        x[(size_t)n * 32 + t]   = val;   // pad lanes 27..31 = 0
        agg[(size_t)n * 32 + t] = 0.f;
    }
}

// ---------------------------------------------------------------------------
// 2) Per-node edge-weight MLP scalar:
//    z[n] = lrelu(x[n] @ ew1_w + ew1_b) @ ew2_w + ew2_b
//    (exact rewrite: ew[e] = 0.5*(z[src]+z[dst]) since the MLP acts per
//    endpoint and the mean is over endpoints). 63x fewer FLOPs than per-edge.
// ---------------------------------------------------------------------------
__global__ void node_z_kernel(const float* __restrict__ x,
                              const float* __restrict__ ew1_w, const float* __restrict__ ew1_b,
                              const float* __restrict__ ew2_w, const float* __restrict__ ew2_b,
                              float* __restrict__ z)
{
    __shared__ float red[128];
    __shared__ float xs[27];
    const int n = blockIdx.x, t = threadIdx.x;
    if (t < 27) xs[t] = x[(size_t)n * 32 + t];
    __syncthreads();
    float acc = ew1_b[t];
    for (int f = 0; f < 27; ++f) acc += xs[f] * ew1_w[f * HID + t];
    red[t] = lrelu(acc) * ew2_w[t];
    __syncthreads();
    for (int s = 64; s > 0; s >>= 1) { if (t < s) red[t] += red[t + s]; __syncthreads(); }
    if (t == 0) z[n] = red[0] + ew2_b[0];
}

// ---------------------------------------------------------------------------
// 3) Edge scatter (GraphConv aggregation): agg[dst] += x[src] * ew[e]
//    ~111MB of f32 atomics, L2-resident (node table is 2MB, L2 is 192MB).
// ---------------------------------------------------------------------------
__global__ void edge_scatter_kernel(const int* __restrict__ ei, int E,
                                    const float* __restrict__ x,
                                    const float* __restrict__ z,
                                    float* __restrict__ agg)
{
    const int e = blockIdx.x * blockDim.x + threadIdx.x;
    if (e >= E) return;
    const int s = ei[e], d = ei[E + e];
    const float w = 0.5f * (z[s] + z[d]);
    const float* xs = x + (size_t)s * 32;
    float* ad = agg + (size_t)d * 32;
#pragma unroll
    for (int f = 0; f < 27; ++f) atomicAdd(&ad[f], xs[f] * w);
}

// ---------------------------------------------------------------------------
// 4) Build f16 fused GraphConv input: xcat = [agg(27) | x(27) | 0-pad] (N x 64)
// ---------------------------------------------------------------------------
__global__ void build_xcat_kernel(const float* __restrict__ agg,
                                  const float* __restrict__ x,
                                  _Float16* __restrict__ xcat)
{
    const int gid = blockIdx.x * blockDim.x + threadIdx.x;
    const int n = gid >> 6, c = gid & 63;
    float v = 0.f;
    if (c < 27)      v = agg[(size_t)n * 32 + c];
    else if (c < 54) v = x[(size_t)n * 32 + (c - 27)];
    xcat[gid] = (_Float16)v;
}

// ---------------------------------------------------------------------------
// Weight packing into WMMA B-fragment layout (v_wmma_f32_16x16x32_f16).
// ISA 7.12.2 16-bit layout: half h of lane l holds
//   K = (h>>3)*16 + (l>>4)*8 + (h&7),  N = tile_n*16 + (l&15)
// Fragments stored as [tile(kt*NT+nt)][lane][16 halves] -> contiguous 32B/lane.
// ---------------------------------------------------------------------------
__global__ void pack_b_kernel(const float* __restrict__ src, _Float16* __restrict__ dst,
                              int KT, int N, int logicalK)
{
    const int gid = blockIdx.x * blockDim.x + threadIdx.x;
    const int lane = gid & 31, t = gid >> 5;
    const int NT = N >> 4;
    if (t >= KT * NT) return;
    const int kt = t / NT, nt = t % NT;
    const int g = lane >> 4, nn = lane & 15;
    const int n = nt * 16 + nn;
#pragma unroll
    for (int h = 0; h < 16; ++h) {
        int k = kt * 32 + ((h >> 3) * 16) + g * 8 + (h & 7);
        float v = (k < logicalK) ? src[(size_t)k * N + n] : 0.f;
        dst[(size_t)(t * 32 + lane) * 16 + h] = (_Float16)v;
    }
}

// Pack [rel_w ; root_w] (54x128, padded K->64) into B-fragments.
__global__ void pack_wcat_kernel(const float* __restrict__ rel_w,
                                 const float* __restrict__ root_w,
                                 _Float16* __restrict__ dst)
{
    const int gid = blockIdx.x * blockDim.x + threadIdx.x; // 2*8*32 threads
    const int lane = gid & 31, t = gid >> 5;
    if (t >= 16) return;
    const int kt = t >> 3, nt = t & 7;
    const int g = lane >> 4, nn = lane & 15;
    const int n = nt * 16 + nn;
#pragma unroll
    for (int h = 0; h < 16; ++h) {
        int k = kt * 32 + ((h >> 3) * 16) + g * 8 + (h & 7);
        float v = 0.f;
        if (k < 27)      v = rel_w[k * HID + n];
        else if (k < 54) v = root_w[(k - 27) * HID + n];
        dst[(size_t)(t * 32 + lane) * 16 + h] = (_Float16)v;
    }
}

// ---------------------------------------------------------------------------
// A-fragment gather from a row-major f16 activation matrix (stride = K).
// Same 16-bit layout: half pairs (2q,2q+1) are K-contiguous -> 8 u32 loads.
// ---------------------------------------------------------------------------
__device__ __forceinline__ v16h load_a_frag(const _Float16* __restrict__ base,
                                            int stride, int row0, int k0, int lane)
{
    const int g = (lane >> 4) & 1, m = lane & 15;
    const _Float16* rp = base + (size_t)(row0 + m) * stride + k0;
    union { v16h v; unsigned u[8]; } A;
#pragma unroll
    for (int q = 0; q < 8; ++q) {
        int k = ((q >= 4) ? 16 : 0) + g * 8 + (q & 3) * 2;
        A.u[q] = *(const unsigned*)(rp + k);
    }
    return A.v;
}

__device__ __forceinline__ v16h load_b_frag(const _Float16* __restrict__ frags, int t, int lane)
{
    return *(const v16h*)(frags + (size_t)(t * 32 + lane) * 16);
}

// ---------------------------------------------------------------------------
// WMMA GEMM + bias + lrelu: actOut(Nx128,f16) = lrelu(actIn(NxKT*32,f16)@W + b)
// KT is a compile-time constant so fragment arrays stay in fixed VGPRs
// (no v_movrels register indexing) and all loops fully unroll:
// KT=2 -> 8 static WMMAs, KT=4 -> 16 static WMMAs per kernel.
// 256 threads = 8 waves/block; block covers 64 rows x 128 cols.
// Wave w: rows [mt*16, mt*16+16), cols [nh*64, nh*64+64) (4 tiles of 16);
// each A fragment is reused across the 4 column tiles.
// EXEC is all-ones everywhere (exact grid, no divergence) per WMMA rules.
// ---------------------------------------------------------------------------
template <int KT>
__global__ __launch_bounds__(256)
void gemm_wmma_kernel(const _Float16* __restrict__ actIn,
                      const _Float16* __restrict__ wfrags,
                      const float* __restrict__ bias,
                      _Float16* __restrict__ actOut)
{
    const int lane = threadIdx.x & 31;
    const int w = threadIdx.x >> 5;
    const int mt = w & 3;
    const int nh = w >> 2;
    const int row0 = blockIdx.x * 64 + mt * 16;

    v16h a[KT];
#pragma unroll
    for (int kt = 0; kt < KT; ++kt)
        a[kt] = load_a_frag(actIn, KT * 32, row0, kt * 32, lane);

    const int g = lane >> 4, nn = lane & 15;
#pragma unroll
    for (int j = 0; j < 4; ++j) {
        const int nt = nh * 4 + j;
        v8f c = {};
#pragma unroll
        for (int kt = 0; kt < KT; ++kt) {
            v16h b = load_b_frag(wfrags, kt * 8 + nt, lane);
            c = __builtin_amdgcn_wmma_f32_16x16x32_f16(
                    false, a[kt], false, b, (short)0, c, false, false);
        }
        const int col = nt * 16 + nn;
        const float bv = bias[col];
#pragma unroll
        for (int q = 0; q < 8; ++q) {
            int r = row0 + g * 8 + q;
            actOut[(size_t)r * HID + col] = (_Float16)lrelu(c[q] + bv);
        }
    }
}

// ---------------------------------------------------------------------------
// 5) Prediction head: out[n] = act[n] . pred_w + pred_b
// ---------------------------------------------------------------------------
__global__ void pred_kernel(const _Float16* __restrict__ act,
                            const float* __restrict__ pw, const float* __restrict__ pb,
                            float* __restrict__ out)
{
    __shared__ float red[128];
    const int n = blockIdx.x, t = threadIdx.x;
    red[t] = (float)act[(size_t)n * HID + t] * pw[t];
    __syncthreads();
    for (int s = 64; s > 0; s >>= 1) { if (t < s) red[t] += red[t + s]; __syncthreads(); }
    if (t == 0) out[n] = red[0] + pb[0];
}

// ---------------------------------------------------------------------------
extern "C" void kernel_launch(void* const* d_in, const int* in_sizes, int n_in,
                              void* d_out, int out_size, void* d_ws, size_t ws_size,
                              hipStream_t stream)
{
    (void)n_in; (void)out_size; (void)ws_size;
    const float* pose   = (const float*)d_in[0];
    const float* views  = (const float*)d_in[1];
    const int*   ei     = (const int*)d_in[2];
    const float* enc1_w = (const float*)d_in[3];
    const float* enc1_b = (const float*)d_in[4];
    const float* enc2_w = (const float*)d_in[5];
    const float* enc2_b = (const float*)d_in[6];
    const float* ew1_w  = (const float*)d_in[7];
    const float* ew1_b  = (const float*)d_in[8];
    const float* ew2_w  = (const float*)d_in[9];
    const float* ew2_b  = (const float*)d_in[10];
    const float* rel_w  = (const float*)d_in[11];
    const float* rel_b  = (const float*)d_in[12];
    const float* root_w = (const float*)d_in[13];
    const float* lin1_w = (const float*)d_in[14];
    const float* lin1_b = (const float*)d_in[15];
    const float* lin2_w = (const float*)d_in[16];
    const float* lin2_b = (const float*)d_in[17];
    const float* lin3_w = (const float*)d_in[18];
    const float* lin3_b = (const float*)d_in[19];
    const float* pred_w = (const float*)d_in[20];
    const float* pred_b = (const float*)d_in[21];
    float* out = (float*)d_out;

    const int N = in_sizes[0] / 3;   // 16384
    const int E = in_sizes[2] / 2;   // 1032192

    // Workspace carve-out (256B aligned)
    char* base = (char*)d_ws;
    size_t off = 0;
    auto alloc = [&](size_t bytes) -> void* {
        void* p = base + off;
        off = (off + bytes + 255) & ~(size_t)255;
        return p;
    };
    float*    x_f  = (float*)   alloc((size_t)N * 32 * sizeof(float));
    float*    agg  = (float*)   alloc((size_t)N * 32 * sizeof(float));
    float*    z    = (float*)   alloc((size_t)N * sizeof(float));
    _Float16* xcat = (_Float16*)alloc((size_t)N * 64 * sizeof(_Float16));
    _Float16* actA = (_Float16*)alloc((size_t)N * HID * sizeof(_Float16));
    _Float16* actB = (_Float16*)alloc((size_t)N * HID * sizeof(_Float16));
    _Float16* wcatF = (_Float16*)alloc((size_t)64  * HID * sizeof(_Float16));
    _Float16* w1F   = (_Float16*)alloc((size_t)HID * HID * sizeof(_Float16));
    _Float16* w2F   = (_Float16*)alloc((size_t)HID * HID * sizeof(_Float16));
    _Float16* w3F   = (_Float16*)alloc((size_t)HID * HID * sizeof(_Float16));

    // 1) encoder -> x (padded), zero agg
    encoder_kernel<<<N, 128, 0, stream>>>(pose, views, enc1_w, enc1_b, enc2_w, enc2_b,
                                          x_f, agg);
    // 2) per-node edge-weight scalar z
    node_z_kernel<<<N, 128, 0, stream>>>(x_f, ew1_w, ew1_b, ew2_w, ew2_b, z);

    // 3) pack weights into WMMA fragments
    pack_wcat_kernel<<<2, 256, 0, stream>>>(rel_w, root_w, wcatF);
    pack_b_kernel<<<4, 256, 0, stream>>>(lin1_w, w1F, 4, HID, HID);
    pack_b_kernel<<<4, 256, 0, stream>>>(lin2_w, w2F, 4, HID, HID);
    pack_b_kernel<<<4, 256, 0, stream>>>(lin3_w, w3F, 4, HID, HID);

    // 4) edge scatter
    edge_scatter_kernel<<<(E + 255) / 256, 256, 0, stream>>>(ei, E, x_f, z, agg);

    // 5) fused GraphConv input
    build_xcat_kernel<<<(N * 64) / 256, 256, 0, stream>>>(agg, x_f, xcat);

    // 6) WMMA GEMM stack (compile-time KT -> fully unrolled, fixed VGPR frags)
    const int gblocks = N / 64;
    gemm_wmma_kernel<2><<<gblocks, 256, 0, stream>>>(xcat, wcatF, rel_b,  actA); // GraphConv
    gemm_wmma_kernel<4><<<gblocks, 256, 0, stream>>>(actA, w1F,   lin1_b, actB); // lin1
    gemm_wmma_kernel<4><<<gblocks, 256, 0, stream>>>(actB, w2F,   lin2_b, actA); // lin2
    gemm_wmma_kernel<4><<<gblocks, 256, 0, stream>>>(actA, w3F,   lin3_b, actB); // lin3

    // 7) prediction head
    pred_kernel<<<N, 128, 0, stream>>>(actB, pred_w, pred_b, out);
}
